// DynamicGRU_47983374631071
// MI455X (gfx1250) — matrix operand order repaired and checked by reading, not compiled, not run on hardware
//
#include <hip/hip_runtime.h>

typedef __attribute__((ext_vector_type(16))) _Float16 v16h;
typedef __attribute__((ext_vector_type(8)))  float    v8f;

#define B_ 4096
#define T_ 200
#define D_ 64
#define H_ 16

__device__ __forceinline__ float fast_rcp(float x) {
#if __has_builtin(__builtin_amdgcn_rcpf)
    return __builtin_amdgcn_rcpf(x);     // v_rcp_f32
#else
    return 1.0f / x;
#endif
}

__device__ __forceinline__ float fast_sigmoid(float x) {
    return fast_rcp(1.0f + __expf(-x));  // v_exp_f32 + v_rcp_f32
}

__device__ __forceinline__ float fast_tanh(float x) {
#if __has_builtin(__builtin_amdgcn_tanhf)
    return __builtin_amdgcn_tanhf(x);    // v_tanh_f32 (gfx1250 TRANS op)
#else
    float xc = fminf(fmaxf(x, -15.0f), 15.0f);
    float e  = __expf(2.0f * xc);
    return (e - 1.0f) * fast_rcp(e + 1.0f);
#endif
}

__device__ __forceinline__ v8f wmma_f16(const v16h& a, const v16h& b, const v8f& c) {
    return __builtin_amdgcn_wmma_f32_16x16x32_f16(
        /*neg_a=*/false, a, /*neg_b=*/false, b,
        /*c_mod=*/(short)0, c, /*reuse_a=*/false, /*reuse_b=*/false);
}

__global__ void __launch_bounds__(32)
DynamicGRU_scan_kernel(const float* __restrict__ x,
                       const float* __restrict__ att,
                       const float* __restrict__ Wr, const float* __restrict__ br,
                       const float* __restrict__ Wu, const float* __restrict__ bu,
                       const float* __restrict__ Wh, const float* __restrict__ bh,
                       float* __restrict__ out)
{
    __shared__ float lds_h[H_ * H_];   // 16x16 row-major staging for C->A transpose
    __shared__ float lds_at[H_];       // per-row attention scalar

    const int lane = threadIdx.x;      // 0..31, wave32
    const int half = lane >> 4;        // 0 or 1
    const int n    = lane & 15;        // C/D column, B column, A row
    const int b0   = blockIdx.x * 16;  // batch tile base

    // ---- Load resident weight B-fragments (f16). B 32x16 layout: K = e + 8*half + (e>=8 ? 8 : 0) ----
    auto load_bfrag = [&](const float* __restrict__ W, int row_base) -> v16h {
        v16h f;
        #pragma unroll
        for (int e = 0; e < 16; ++e) {
            int k = e + 8 * half + (e >= 8 ? 8 : 0);
            f[e] = (_Float16)W[(row_base + k) * H_ + n];
        }
        return f;
    };
    // h-part fragments: only K<16 valid (K=16..31 zero-padded)
    auto load_bfrag_pad = [&](const float* __restrict__ W, int row_base) -> v16h {
        v16h f;
        #pragma unroll
        for (int e = 0; e < 16; ++e) {
            int k = e + 8 * half + (e >= 8 ? 8 : 0);
            f[e] = (k < 16) ? (_Float16)W[(row_base + k) * H_ + n] : (_Float16)0.0f;
        }
        return f;
    };

    // cat = [x, h]: Wr/Wu rows 0..63 -> x, rows 64..79 -> h
    const v16h Br0 = load_bfrag(Wr, 0),  Br1 = load_bfrag(Wr, 32), Brh = load_bfrag_pad(Wr, 64);
    const v16h Bu0 = load_bfrag(Wu, 0),  Bu1 = load_bfrag(Wu, 32), Buh = load_bfrag_pad(Wu, 64);
    // hh-gate cat = [r*h, x]: Wh rows 0..15 -> r*h, rows 16..79 -> x
    const v16h Bh0 = load_bfrag(Wh, 16), Bh1 = load_bfrag(Wh, 48), Bhh = load_bfrag_pad(Wh, 0);

    const float cbr = br[n], cbu = bu[n], cbh = bh[n];

    // ---- A-fragment loader for x. A 16x32 f16: K = e + 8*half + (e>=8 ? 8 : 0) ----
    const float* __restrict__ xr = x + (size_t)(b0 + n) * (T_ * D_);
    auto load_x_frag = [&](const float* __restrict__ xt, int ko) -> v16h {
        const float* p0 = xt + ko + 8 * half;        // K run: e=0..7
        const float* p1 = p0 + 16;                   // K run: e=8..15
        float4 a0 = *reinterpret_cast<const float4*>(p0);
        float4 a1 = *reinterpret_cast<const float4*>(p0 + 4);
        float4 c0 = *reinterpret_cast<const float4*>(p1);
        float4 c1 = *reinterpret_cast<const float4*>(p1 + 4);
        v16h f;
        f[0]  = (_Float16)a0.x; f[1]  = (_Float16)a0.y; f[2]  = (_Float16)a0.z; f[3]  = (_Float16)a0.w;
        f[4]  = (_Float16)a1.x; f[5]  = (_Float16)a1.y; f[6]  = (_Float16)a1.z; f[7]  = (_Float16)a1.w;
        f[8]  = (_Float16)c0.x; f[9]  = (_Float16)c0.y; f[10] = (_Float16)c0.z; f[11] = (_Float16)c0.w;
        f[12] = (_Float16)c1.x; f[13] = (_Float16)c1.y; f[14] = (_Float16)c1.z; f[15] = (_Float16)c1.w;
        return f;
    };

    // C/D layout: lane holds rows M = i + 8*half (i=0..7), column N = n
    auto c_to_lds = [&](const v8f& c) {
        #pragma unroll
        for (int i = 0; i < 8; ++i)
            lds_h[(i + 8 * half) * H_ + n] = c[i];
    };
    // Read back 16x16 as zero-padded 16x32 A fragment: valid K = e + 8*half for e<8
    auto lds_to_afrag = [&]() -> v16h {
        v16h f;
        #pragma unroll
        for (int e = 0; e < 8; ++e)
            f[e] = (_Float16)lds_h[n * H_ + (e + 8 * half)];
        #pragma unroll
        for (int e = 8; e < 16; ++e)
            f[e] = (_Float16)0.0f;
        return f;
    };

    // ---- Recurrent state ----
    v8f h_c;                      // fp32, C layout
    v16h a_h;                     // f16, A layout (padded), for the h-part WMMAs
    #pragma unroll
    for (int i = 0; i < 8; ++i)  h_c[i] = 0.0f;
    #pragma unroll
    for (int e = 0; e < 16; ++e) a_h[e] = (_Float16)0.0f;

    for (int t = 0; t < T_; ++t) {
        const float* xt = xr + (size_t)t * D_;
        const v16h ax0 = load_x_frag(xt, 0);
        const v16h ax1 = load_x_frag(xt, 32);
        if (t + 1 < T_) __builtin_prefetch(xt + D_, 0, 1);   // global_prefetch_b8

        const float at_val = att[(size_t)(b0 + n) * T_ + t];

        // r and u gates: bias + x-part (K=64) + h-part (K=16 padded)
        v8f accr, accu;
        #pragma unroll
        for (int i = 0; i < 8; ++i) { accr[i] = cbr; accu[i] = cbu; }
        accr = wmma_f16(ax0, Br0, accr);
        accu = wmma_f16(ax0, Bu0, accu);
        accr = wmma_f16(ax1, Br1, accr);
        accu = wmma_f16(ax1, Bu1, accu);
        accr = wmma_f16(a_h, Brh, accr);
        accu = wmma_f16(a_h, Buh, accu);

        // r*h in C layout, stage to LDS for use as A operand
        v8f rh;
        #pragma unroll
        for (int i = 0; i < 8; ++i)
            rh[i] = fast_sigmoid(accr[i]) * h_c[i];

        __syncthreads();            // protect a_h reads of previous iteration
        c_to_lds(rh);
        lds_at[n] = at_val;         // both halves write identical value: benign
        __syncthreads();
        const v16h arh = lds_to_afrag();
        float atv[8];
        #pragma unroll
        for (int i = 0; i < 8; ++i) atv[i] = lds_at[i + 8 * half];

        // candidate gate
        v8f acch;
        #pragma unroll
        for (int i = 0; i < 8; ++i) acch[i] = cbh;
        acch = wmma_f16(ax0, Bh0, acch);
        acch = wmma_f16(ax1, Bh1, acch);
        acch = wmma_f16(arh, Bhh, acch);

        // h = (1 - a*u)*h + a*u*tanh(acch)
        #pragma unroll
        for (int i = 0; i < 8; ++i) {
            float uu = atv[i] * fast_sigmoid(accu[i]);
            float hh = fast_tanh(acch[i]);
            h_c[i] = fmaf(uu, hh - h_c[i], h_c[i]);
        }

        // refresh h A-fragment for next step
        __syncthreads();
        c_to_lds(h_c);
        __syncthreads();
        a_h = lds_to_afrag();
    }

    // write h_final [B,16]
    #pragma unroll
    for (int i = 0; i < 8; ++i)
        out[(size_t)(b0 + i + 8 * half) * H_ + n] = h_c[i];
}

extern "C" void kernel_launch(void* const* d_in, const int* in_sizes, int n_in,
                              void* d_out, int out_size, void* d_ws, size_t ws_size,
                              hipStream_t stream) {
    const float* x   = (const float*)d_in[0];
    const float* att = (const float*)d_in[1];
    const float* Wr  = (const float*)d_in[2];
    const float* br  = (const float*)d_in[3];
    const float* Wu  = (const float*)d_in[4];
    const float* bu  = (const float*)d_in[5];
    const float* Wh  = (const float*)d_in[6];
    const float* bh  = (const float*)d_in[7];

    dim3 grid(B_ / 16);   // 256 tiles, one wave32 each
    dim3 block(32);
    DynamicGRU_scan_kernel<<<grid, block, 0, stream>>>(
        x, att, Wr, br, Wu, bu, Wh, bh, (float*)d_out);
}
